// SelfAttention_35424890257576
// MI455X (gfx1250) — matrix operand order repaired and checked by reading, not compiled, not run on hardware
//
#include <hip/hip_runtime.h>

// ---------------------------------------------------------------------------
// Fused self-attention for MI455X (gfx1250, wave32, WMMA bf16 16x16x32).
//   B=4, H=W=64 -> N=4096, C=256, C8=32, GAMMA=0.01
// Pipeline:
//   prep_kernel : transpose weights to bf16 (wfT/wgT/whT [32][256], wvT [256][32])
//   proj_kernel : f = x@wf+bf, g = x@wg+bg (row-major bf16), h -> hT [B][32][N]
//   attn_kernel : flash attention (online softmax) + (GAMMA*ctx)@wv + bv + x
// ---------------------------------------------------------------------------

typedef __attribute__((ext_vector_type(16))) __bf16 v16bf;
typedef __attribute__((ext_vector_type(8)))  __bf16 v8bf;
typedef __attribute__((ext_vector_type(8)))  float  v8f;

#define NN   4096
#define CC   256
#define C8   32
#define LOG2E 1.4426950408889634f

// Native bf16 conversion: lowers to gfx1250 hardware cvt (packed when paired).
static __device__ __forceinline__ __bf16 f2bf(float f) { return (__bf16)f; }
static __device__ __forceinline__ unsigned short f2bfu(float f) {
  return __builtin_bit_cast(unsigned short, (__bf16)f);
}

static __device__ __forceinline__ v16bf ldb16(const unsigned short* p0,
                                              const unsigned short* p1) {
  v8bf a = *(const v8bf*)p0;
  v8bf b = *(const v8bf*)p1;
  v16bf r;
#pragma unroll
  for (int i = 0; i < 8; ++i) { r[i] = a[i]; r[i + 8] = b[i]; }
  return r;
}

// RA = reuse-A hint: set on WMMAs whose A operand matches the previous WMMA.
static __device__ __forceinline__ v8f wmma_bf16(v16bf a, v16bf b, v8f c) {
  return __builtin_amdgcn_wmma_f32_16x16x32_bf16(false, a, false, b,
                                                 (short)0, c, false, false);
}
static __device__ __forceinline__ v8f wmma_bf16_ra(v16bf a, v16bf b, v8f c) {
  return __builtin_amdgcn_wmma_f32_16x16x32_bf16(false, a, false, b,
                                                 (short)0, c, true, false);
}

// ds_swizzle group-of-32 xor patterns within 16-lane halves (and=0x1F, or=0)
#define SWZ(x, imm) \
  __builtin_bit_cast(float, __builtin_amdgcn_ds_swizzle(__builtin_bit_cast(int, (x)), (imm)))

static __device__ __forceinline__ float rowmax16(float x) {
  x = fmaxf(x, SWZ(x, 0x041F));
  x = fmaxf(x, SWZ(x, 0x081F));
  x = fmaxf(x, SWZ(x, 0x101F));
  x = fmaxf(x, SWZ(x, 0x201F));
  return x;
}
static __device__ __forceinline__ float rowsum16(float x) {
  x += SWZ(x, 0x041F);
  x += SWZ(x, 0x081F);
  x += SWZ(x, 0x101F);
  x += SWZ(x, 0x201F);
  return x;
}

// ---------------------------------------------------------------------------
// Weight transpose / bf16 conversion.  wfT/wgT/whT: [C8][C], wvT: [C][C8]
// ---------------------------------------------------------------------------
__global__ __launch_bounds__(256) void prep_kernel(
    const float* __restrict__ wf, const float* __restrict__ wg,
    const float* __restrict__ wh, const float* __restrict__ wv,
    unsigned short* __restrict__ wfT, unsigned short* __restrict__ wgT,
    unsigned short* __restrict__ whT, unsigned short* __restrict__ wvT) {
  int id = blockIdx.x * 256 + threadIdx.x;    // 0 .. 32767
  int m  = id >> 13;                          // which matrix
  int i  = id & 8191;
  if (m < 3) {
    const float* w       = (m == 0) ? wf : (m == 1) ? wg : wh;
    unsigned short* wT   = (m == 0) ? wfT : (m == 1) ? wgT : whT;
    int c8 = i >> 8, c = i & 255;
    wT[c8 * CC + c] = f2bfu(w[c * C8 + c8]);
  } else {
    int c = i >> 5, c8 = i & 31;
    wvT[c * C8 + c8] = f2bfu(wv[c8 * CC + c]);
  }
}

// ---------------------------------------------------------------------------
// Projections: each wave computes a 16-row tile of f,g,h via WMMA (K=256).
//   fws/gws: [B*N][32] bf16 row-major;  hT: [B][32][N] bf16 (transposed)
// ---------------------------------------------------------------------------
__global__ __launch_bounds__(256) void proj_kernel(
    const float* __restrict__ x,
    const float* __restrict__ bfv, const float* __restrict__ bgv,
    const float* __restrict__ bhv,
    const unsigned short* __restrict__ wfT, const unsigned short* __restrict__ wgT,
    const unsigned short* __restrict__ whT,
    unsigned short* __restrict__ fws, unsigned short* __restrict__ gws,
    unsigned short* __restrict__ hT) {
  const int lane = threadIdx.x & 31;
  const int wave = threadIdx.x >> 5;
  const int wt   = blockIdx.x * 8 + wave;   // 1024 tiles of 16 rows (B*N/16)
  const int r0   = wt * 16;
  const int hl   = lane >> 4;
  const int ln   = lane & 15;
  const int akb  = hl * 16;                 // A K-base per half-wave
  const int bkb  = hl * 16;                 // B K-base per half-wave

  v8f aF0 = {}, aF1 = {}, aG0 = {}, aG1 = {}, aH0 = {}, aH1 = {};

  const float* xr = x + (long)(r0 + ln) * CC;
  const unsigned short* pf0 = wfT + (long)ln * CC;
  const unsigned short* pf1 = wfT + (long)(16 + ln) * CC;
  const unsigned short* pg0 = wgT + (long)ln * CC;
  const unsigned short* pg1 = wgT + (long)(16 + ln) * CC;
  const unsigned short* ph0 = whT + (long)ln * CC;
  const unsigned short* ph1 = whT + (long)(16 + ln) * CC;

#pragma unroll
  for (int kb = 0; kb < CC; kb += 32) {
    v8f xlo = *(const v8f*)(xr + kb + akb);
    v8f xhi = *(const v8f*)(xr + kb + akb + 8);
    v16bf A;
#pragma unroll
    for (int i = 0; i < 8; ++i) { A[i] = f2bf(xlo[i]); A[i + 8] = f2bf(xhi[i]); }

    v16bf Bf0 = ldb16(pf0 + kb + bkb, pf0 + kb + bkb + 8);
    v16bf Bf1 = ldb16(pf1 + kb + bkb, pf1 + kb + bkb + 8);
    v16bf Bg0 = ldb16(pg0 + kb + bkb, pg0 + kb + bkb + 8);
    v16bf Bg1 = ldb16(pg1 + kb + bkb, pg1 + kb + bkb + 8);
    v16bf Bh0 = ldb16(ph0 + kb + bkb, ph0 + kb + bkb + 8);
    v16bf Bh1 = ldb16(ph1 + kb + bkb, ph1 + kb + bkb + 8);

    // 6-WMMA chain sharing A: reuse-A hint on the repeated members
    aF0 = wmma_bf16(A, Bf0, aF0);
    aF1 = wmma_bf16_ra(A, Bf1, aF1);
    aG0 = wmma_bf16_ra(A, Bg0, aG0);
    aG1 = wmma_bf16_ra(A, Bg1, aG1);
    aH0 = wmma_bf16_ra(A, Bh0, aH0);
    aH1 = wmma_bf16_ra(A, Bh1, aH1);
  }

  const float bf0 = bfv[ln],      bf1 = bfv[16 + ln];
  const float bg0 = bgv[ln],      bg1 = bgv[16 + ln];
  const float bh0 = bhv[ln],      bh1 = bhv[16 + ln];

#pragma unroll
  for (int r = 0; r < 8; ++r) {
    int  grow = r0 + r + 8 * hl;            // flat row over B*N
    long ro   = (long)grow * C8;
    int  b    = grow >> 12;                 // /4096
    int  nI   = grow & 4095;
    fws[ro + ln]      = f2bfu(aF0[r] + bf0);
    fws[ro + 16 + ln] = f2bfu(aF1[r] + bf1);
    gws[ro + ln]      = f2bfu(aG0[r] + bg0);
    gws[ro + 16 + ln] = f2bfu(aG1[r] + bg1);
    hT[(long)(b * C8 + ln)      * NN + nI] = f2bfu(aH0[r] + bh0);
    hT[(long)(b * C8 + 16 + ln) * NN + nI] = f2bfu(aH1[r] + bh1);
  }
}

// ---------------------------------------------------------------------------
// Flash attention + output projection + residual.
// 128 blocks x 256 threads; each wave owns 16 query rows of one batch.
// ---------------------------------------------------------------------------
__global__ __launch_bounds__(256) void attn_kernel(
    const float* __restrict__ x, const float* __restrict__ bv,
    const unsigned short* __restrict__ fws, const unsigned short* __restrict__ gws,
    const unsigned short* __restrict__ hT, const unsigned short* __restrict__ wvT,
    float* __restrict__ out) {
  __shared__ unsigned short lds[8][16 * 40];   // per-wave 16x32 bf16 tile, padded

  const int lane = threadIdx.x & 31;
  const int wave = threadIdx.x >> 5;
  const int b    = blockIdx.x >> 5;                       // 32 blocks per batch
  const int qb   = (blockIdx.x & 31) * 128 + wave * 16;   // query base in batch
  const int hl   = lane >> 4;
  const int ln   = lane & 15;
  unsigned short* P = &lds[wave][0];

  const long base32 = (long)b * NN * C8;
  const unsigned short* gb = gws + base32;
  const unsigned short* hb = hT + (long)b * C8 * NN;

  // A-matrix: f tile, rows qb..qb+15 (loaded once)
  const unsigned short* fr = fws + base32 + (long)(qb + ln) * C8 + hl * 16;
  v16bf fA = ldb16(fr, fr + 8);

  float mrun[8], lpart[8];                 // lpart: per-lane partial row sums
  v8f accLo = {}, accHi = {};
#pragma unroll
  for (int r = 0; r < 8; ++r) { mrun[r] = -3.0e38f; lpart[r] = 0.0f; }

  for (int j = 0; j < NN; j += 32) {
    if (j + 32 < NN) {   // prefetch next key chunk -> global_prefetch_b8
      __builtin_prefetch(gb + (long)(j + 32 + ln) * C8, 0, 0);
      __builtin_prefetch(hb + (long)ln * NN + j + 32, 0, 0);
    }
    // s = f @ g^T for 32 keys: two 16x16 tiles (shared A -> reuse hint)
    const unsigned short* g0 = gb + (long)(j + ln) * C8 + hl * 16;
    const unsigned short* g1 = gb + (long)(j + 16 + ln) * C8 + hl * 16;
    v16bf gB0 = ldb16(g0, g0 + 8);
    v16bf gB1 = ldb16(g1, g1 + 8);
    v8f z = {};
    v8f s0 = wmma_bf16(fA, gB0, z);
    v8f s1 = wmma_bf16_ra(fA, gB1, z);

    // online softmax (row r of accumulator == query row r + 8*hl).
    // Row sums are kept as per-lane partials (scale is uniform per 16-lane
    // group, so the cross-lane reduction commutes to after the loop).
    v8f p0, p1;
#pragma unroll
    for (int r = 0; r < 8; ++r) {
      float tmax  = rowmax16(fmaxf(s0[r], s1[r]));
      float mnew  = fmaxf(mrun[r], tmax);
      float scale = __builtin_amdgcn_exp2f((mrun[r] - mnew) * LOG2E);
      mrun[r] = mnew;
      float e0 = __builtin_amdgcn_exp2f((s0[r] - mnew) * LOG2E);
      float e1 = __builtin_amdgcn_exp2f((s1[r] - mnew) * LOG2E);
      p0[r] = e0;
      p1[r] = e1;
      lpart[r] = lpart[r] * scale + (e0 + e1);
      accLo[r] *= scale;
      accHi[r] *= scale;
    }

    // Stage P (16x32 bf16) through LDS: C/D layout -> A layout transpose
#pragma unroll
    for (int r = 0; r < 8; ++r) {
      int row = r + 8 * hl;
      P[row * 40 + ln]      = f2bfu(p0[r]);
      P[row * 40 + 16 + ln] = f2bfu(p1[r]);
    }
    __asm__ volatile("s_wait_dscnt 0x0" ::: "memory");
    const unsigned short* pr = P + ln * 40 + hl * 16;
    v16bf pA = ldb16(pr, pr + 8);

    // ctx += P @ h (h^T rows are contiguous -> clean B loads; shared A)
    const unsigned short* h0 = hb + (long)ln * NN + j + hl * 16;
    const unsigned short* h1 = hb + (long)(16 + ln) * NN + j + hl * 16;
    v16bf hB0 = ldb16(h0, h0 + 8);
    v16bf hB1 = ldb16(h1, h1 + 8);
    accLo = wmma_bf16(pA, hB0, accLo);
    accHi = wmma_bf16_ra(pA, hB1, accHi);
  }

  // finalize row sums (single cross-lane reduction), normalize, apply GAMMA,
  // stage ctx through LDS for the output WMMA
#pragma unroll
  for (int r = 0; r < 8; ++r) {
    int   row = r + 8 * hl;
    float lr  = rowsum16(lpart[r]);
    float inv = 0.01f / lr;
    P[row * 40 + ln]      = f2bfu(accLo[r] * inv);
    P[row * 40 + 16 + ln] = f2bfu(accHi[r] * inv);
  }
  __asm__ volatile("s_wait_dscnt 0x0" ::: "memory");
  const unsigned short* cr = P + ln * 40 + hl * 16;
  v16bf cA = ldb16(cr, cr + 8);

  // out = ctx @ wv + bv + x   (16 column-tiles of 16)
  for (int t = 0; t < 16; ++t) {
    int col = t * 16 + ln;
    const unsigned short* wv0 = wvT + (long)col * C8 + hl * 16;
    v16bf wB = ldb16(wv0, wv0 + 8);
    v8f z = {};
    v8f d = wmma_bf16(cA, wB, z);
    float bvc = bv[col];
#pragma unroll
    for (int r = 0; r < 8; ++r) {
      int  grow = qb + r + 8 * hl;
      long oi   = ((long)b * NN + grow) * CC + col;
      out[oi] = d[r] + bvc + x[oi];
    }
  }
}

// ---------------------------------------------------------------------------
extern "C" void kernel_launch(void* const* d_in, const int* in_sizes, int n_in,
                              void* d_out, int out_size, void* d_ws, size_t ws_size,
                              hipStream_t stream) {
  const float* x  = (const float*)d_in[0];
  const float* wf = (const float*)d_in[1];
  const float* bf = (const float*)d_in[2];
  const float* wg = (const float*)d_in[3];
  const float* bg = (const float*)d_in[4];
  const float* wh = (const float*)d_in[5];
  const float* bh = (const float*)d_in[6];
  const float* wv = (const float*)d_in[7];
  const float* bv = (const float*)d_in[8];
  float* out = (float*)d_out;

  // workspace layout (bf16 stored as ushort)
  unsigned short* fws = (unsigned short*)d_ws;          // 4*4096*32
  unsigned short* gws = fws + 4 * NN * C8;              // 4*4096*32
  unsigned short* hT  = gws + 4 * NN * C8;              // 4*32*4096
  unsigned short* wfT = hT  + 4 * NN * C8;              // 32*256
  unsigned short* wgT = wfT + C8 * CC;
  unsigned short* whT = wgT + C8 * CC;
  unsigned short* wvT = whT + C8 * CC;                  // 256*32

  prep_kernel<<<128, 256, 0, stream>>>(wf, wg, wh, wv, wfT, wgT, whT, wvT);
  proj_kernel<<<128, 256, 0, stream>>>(x, bf, bg, bh, wfT, wgT, whT, fws, gws, hT);
  attn_kernel<<<128, 256, 0, stream>>>(x, bv, fws, gws, hT, wvT, out);
}
